// Correlation_13898514170221
// MI455X (gfx1250) — matrix operand order repaired and checked by reading, not compiled, not run on hardware
//
#include <hip/hip_runtime.h>

// ---- types -----------------------------------------------------------------
typedef __attribute__((ext_vector_type(16))) _Float16 v16h;
typedef __attribute__((ext_vector_type(8)))  _Float16 v8h;
typedef __attribute__((ext_vector_type(2)))  _Float16 h2;
typedef __attribute__((ext_vector_type(8)))  float    v8f;

#define C_CH     128
#define HW       160
#define PLANE    (HW * HW)          // 25600
#define DISP     4
#define PATCH    9                  // 2*DISP+1
#define NSHIFT   81
#define HALO_W   24                 // 16 + 2*DISP
#define HALO_PX  (PATCH * HALO_W)   // 216
#define TILE_N   16
#define PXSTRIDE 136                // halfs per LDS pixel (272B): lane n -> banks 4n..4n+3
#define NPX      (HALO_PX + TILE_N) // 232 pixels staged
#define MTILES   14                 // ceil(216/16)
#define KSTEPS   4                  // 128 / 32

static __device__ inline v16h cat16(v8h lo, v8h hi) {
    return __builtin_shufflevector(lo, hi, 0,1,2,3,4,5,6,7,8,9,10,11,12,13,14,15);
}

__global__ __launch_bounds__(128)
void corr_wmma_kernel(const float* __restrict__ x1,
                      const float* __restrict__ x2,
                      float* __restrict__ out) {
    __shared__ _Float16 smem[NPX * PXSTRIDE];   // 63104 bytes

    const int tid  = threadIdx.x;
    const int wave = tid >> 5;
    const int lane = tid & 31;
    const int n    = lane & 15;          // N (x1 pixel) and A-row M index
    const int hiH  = lane >> 4;          // lane-half selector

    const int b  = blockIdx.z;
    const int h  = blockIdx.y;
    const int w0 = blockIdx.x * TILE_N;

    const float* x1b = x1 + (size_t)b * C_CH * PLANE;
    const float* x2b = x2 + (size_t)b * C_CH * PLANE;

    // ---- stage x2 halo (9 rows x 24 cols x 128 ch) into LDS as f16 [px][ch]
    // channel-pair outer (split over waves), halo-row inner (unrolled):
    // row pointer is scalar-uniform -> saddr-form b32 loads; DS stores use
    // immediate offsets (r*HALO_W*PXSTRIDE*2 <= 52 KB).
    {
        const int  wc  = w0 - DISP + lane;
        const bool uok = lane < HALO_W;
        const bool wok = uok && ((unsigned)wc < (unsigned)HW);
        for (int cp = wave; cp < C_CH / 2; cp += 4) {
            const int c0 = cp << 1;
            const float* p0 = x2b + (size_t)c0 * PLANE;           // uniform
            _Float16*    sp = smem + lane * PXSTRIDE + c0;        // per-lane base
#pragma unroll
            for (int r = 0; r < PATCH; ++r) {
                const int hr = h - DISP + r;
                float f0 = 0.0f, f1 = 0.0f;
                if (wok && (unsigned)hr < (unsigned)HW) {
                    const float* pr = p0 + (size_t)hr * HW;       // uniform
                    f0 = pr[wc];
                    f1 = pr[wc + PLANE];
                }
                if (uok) {
                    h2 pk; pk.x = (_Float16)f0; pk.y = (_Float16)f1;
                    *(h2*)(sp + r * HALO_W * PXSTRIDE) = pk;
                }
            }
        }
    }
    // ---- stage x1 tile (16 px x 128 ch) at pixel slots 216..231
    for (int cp = wave; cp < C_CH / 2; cp += 4) {
        const int c0 = cp << 1;
        const float* pr = x1b + (size_t)c0 * PLANE + (size_t)h * HW;  // uniform
        if (lane < TILE_N) {
            h2 pk;
            pk.x = (_Float16)pr[w0 + lane];
            pk.y = (_Float16)pr[w0 + lane + PLANE];
            *(h2*)(smem + (HALO_PX + lane) * PXSTRIDE + c0) = pk;
        }
    }
    __syncthreads();

    // ---- B fragments (x1 pixels): lanes 0-15 K {0..7,16..23}, lanes 16-31 K {8..15,24..31}
    v16h Bf[KSTEPS];
#pragma unroll
    for (int k = 0; k < KSTEPS; ++k) {
        const _Float16* bp = smem + (HALO_PX + n) * PXSTRIDE + 32 * k + hiH * 8;
        Bf[k] = cat16(*(const v8h*)bp, *(const v8h*)(bp + 16));
    }

    const float scale = 1.0f / (float)(C_CH * C_CH);   // mean over c, then /c again

    // uniform output base: out + b*81*PLANE + h*HW + w0  (per-lane += n + shift*PLANE)
    float* outp = out + (size_t)b * NSHIFT * PLANE + (size_t)h * HW + w0;

    // ---- Gram band: wave takes M-tiles wave, wave+4, ...
    for (int t = wave; t < MTILES; t += 4) {
        v8f acc = {};
#pragma unroll
        for (int k = 0; k < KSTEPS; ++k) {
            // A 16x32: M = lane&15; lane<16 K {0..7,16..23}, lane>=16 K {8..15,24..31}
            const _Float16* ap = smem + (16 * t + n) * PXSTRIDE + 32 * k + hiH * 8;
            const v16h Af = cat16(*(const v8h*)ap, *(const v8h*)(ap + 16));
            acc = __builtin_amdgcn_wmma_f32_16x16x32_f16(
                false, Af, false, Bf[k], (short)0, acc, false, false);
        }
        // ---- band extraction: D VGPR v -> M = 16t + v + 8*hiH, N = n
        const unsigned m0 = 16u * t + 8u * hiH;
#pragma unroll
        for (int v = 0; v < 8; ++v) {
            const unsigned m = m0 + v;
            const unsigned r = m / (unsigned)HALO_W;        // di (0..8)
            const unsigned u = m - r * (unsigned)HALO_W;    // halo col (0..23)
            const unsigned j = u - (unsigned)n;             // dj, valid iff <= 8
            if ((m < (unsigned)HALO_PX) & (j <= 8u)) {
                const unsigned idx = (r * (unsigned)PATCH + j) * (unsigned)PLANE
                                   + (unsigned)n;
                outp[idx] = acc[v] * scale;
            }
        }
    }
}

extern "C" void kernel_launch(void* const* d_in, const int* in_sizes, int n_in,
                              void* d_out, int out_size, void* d_ws, size_t ws_size,
                              hipStream_t stream) {
    (void)in_sizes; (void)n_in; (void)d_ws; (void)ws_size; (void)out_size;
    const float* x1 = (const float*)d_in[0];
    const float* x2 = (const float*)d_in[1];
    float* out = (float*)d_out;
    dim3 grid(HW / TILE_N, HW, 8);   // (10, 160, 8)
    dim3 block(128);                 // 4 waves share one halo
    corr_wmma_kernel<<<grid, block, 0, stream>>>(x1, x2, out);
}
